// ViT_36773509988478
// MI455X (gfx1250) — compile-verified, matrix operand order checked by reading, not run on hardware
//
#include <hip/hip_runtime.h>
#include <hip/hip_bf16.h>
#include <math.h>

typedef __attribute__((ext_vector_type(16))) _Float16 v16h;
typedef __attribute__((ext_vector_type(2)))  __fp16   v2fp;
typedef __attribute__((ext_vector_type(8)))  float    v8f;

#define B_    32
#define NP    1024
#define IND   8
#define D_    64
#define H_    8
#define DH_   8
#define S_    1025
#define OUTD  9
#define SPAD  1056          // 33 * 32
#define NCHUNK 33
#define QTILES 65           // ceil(1025/16)
#define QKSTRIDE 16         // q/k rows stored padded to 16 halfs (upper 8 = 0)
#define EPSF  1e-5f

// Interleaved position of key (kk in 0..31) inside a 32-key chunk so that the
// P-tile C-layout -> A-layout LDS pass uses packed 32-bit stores and a single
// contiguous 32B read per lane. Pure permutation of the PV contraction index;
// V is stored pre-permuted to match.
__device__ __host__ __forceinline__ int kpos(int kk) {
  int grp = kk >> 3;                 // 0:0-7 1:8-15 2:16-23 3:24-31
  int off = kk & 7;
  return ((grp & 1) << 4) | (off << 1) | (grp >> 1);
}

// ---------------- Kernel 0: zero page for padded WMMA operand loads ----------------
__global__ void k_zero(_Float16* z) { z[threadIdx.x] = (_Float16)0.f; }

// ---------------- Kernel 1: patch embed + cls + positional encoding ----------------
__global__ void k_tokens(const float* __restrict__ images, const float* __restrict__ w_map,
                         const float* __restrict__ b_map, const float* __restrict__ cls,
                         float* __restrict__ tokens) {
  int idx = blockIdx.x * blockDim.x + threadIdx.x;
  if (idx >= B_ * S_ * D_) return;
  int d = idx % D_;
  int s = (idx / D_) % S_;
  int b = idx / (D_ * S_);
  float t;
  if (s == 0) {
    t = cls[d];
  } else {
    const float* p = images + ((size_t)b * NP + (s - 1)) * IND;
    float acc = b_map[d];
#pragma unroll
    for (int j = 0; j < IND; ++j) acc += p[j] * w_map[j * D_ + d];
    t = acc;
  }
  float fs = (float)s;
  float pe;
  if ((d & 1) == 0) {
    pe = __sinf(fs * __powf(10000.0f, -(float)d / 64.0f));
  } else {
    // reference: cos(i / (10000^(j-1) / d)) == cos(i * d / 10000^(j-1))
    pe = __cosf(fs * 64.0f * __powf(10000.0f, -(float)(d - 1)));
  }
  tokens[idx] = t + pe;
}

// ---------------- Kernel 2: per-batch mean / rsqrt(var) over S*D ----------------
__global__ void k_stats(const float* __restrict__ x, float* __restrict__ mu, float* __restrict__ rs) {
  int b = blockIdx.x;
  const float* p = x + (size_t)b * S_ * D_;
  float s = 0.f, s2 = 0.f;
  for (int i = threadIdx.x; i < S_ * D_; i += blockDim.x) {
    float v = p[i];
    s += v; s2 += v * v;
  }
  __shared__ float sh[256], sh2[256];
  sh[threadIdx.x] = s; sh2[threadIdx.x] = s2;
  __syncthreads();
  for (int off = 128; off > 0; off >>= 1) {
    if ((int)threadIdx.x < off) { sh[threadIdx.x] += sh[threadIdx.x + off]; sh2[threadIdx.x] += sh2[threadIdx.x + off]; }
    __syncthreads();
  }
  if (threadIdx.x == 0) {
    float N = (float)(S_ * D_);
    float m = sh[0] / N;
    float var = sh2[0] / N - m * m;
    mu[b] = m;
    rs[b] = rsqrtf(var + EPSF);
  }
}

// ---------------- Kernel 3: LN1 + QKV projection, f16; q/k padded to 16, V transposed+permuted ----------------
// 1/sqrt(DH) folded into q so the attention loop needs no score scaling.
__global__ void k_qkv(const float* __restrict__ tokens, const float* __restrict__ mu, const float* __restrict__ rs,
                      const float* __restrict__ g1, const float* __restrict__ be1,
                      const float* __restrict__ wq, const float* __restrict__ bq,
                      const float* __restrict__ wk, const float* __restrict__ bk,
                      const float* __restrict__ wv, const float* __restrict__ bv,
                      _Float16* __restrict__ qo, _Float16* __restrict__ ko, _Float16* __restrict__ vT) {
  int idx = blockIdx.x * blockDim.x + threadIdx.x;
  if (idx >= B_ * H_ * SPAD) return;
  int s = idx % SPAD;
  int h = (idx / SPAD) % H_;
  int b = idx / (SPAD * H_);
  size_t rbase = ((size_t)(b * H_ + h) * SPAD + s) * QKSTRIDE;
  // V stored transposed [dh][key], keys permuted within each 32-chunk by kpos()
  size_t vpos = (size_t)(s & ~31) + kpos(s & 31);
  size_t vbase = ((size_t)(b * H_ + h) * DH_) * SPAD + vpos;
  // zero the pad halves (and full rows for s >= S)
#pragma unroll
  for (int e = DH_; e < QKSTRIDE; ++e) { qo[rbase + e] = (_Float16)0.f; ko[rbase + e] = (_Float16)0.f; }
  if (s >= S_) {
#pragma unroll
    for (int e = 0; e < DH_; ++e) {
      qo[rbase + e] = (_Float16)0.f;
      ko[rbase + e] = (_Float16)0.f;
      vT[vbase + (size_t)e * SPAD] = (_Float16)0.f;
    }
    return;
  }
  float m = mu[b], r = rs[b];
  float xn[DH_];
#pragma unroll
  for (int j = 0; j < DH_; ++j) {
    int d = h * DH_ + j;
    float x = tokens[((size_t)b * S_ + s) * D_ + d];
    xn[j] = (x - m) * r * g1[s * D_ + d] + be1[s * D_ + d];
  }
  const float qscale = 0.35355339059327379f;   // 1/sqrt(DH)
#pragma unroll
  for (int e = 0; e < DH_; ++e) {
    float aq = bq[h * DH_ + e], ak = bk[h * DH_ + e], av = bv[h * DH_ + e];
#pragma unroll
    for (int dd = 0; dd < DH_; ++dd) {
      float x = xn[dd];
      aq += x * wq[(h * DH_ + dd) * DH_ + e];
      ak += x * wk[(h * DH_ + dd) * DH_ + e];
      av += x * wv[(h * DH_ + dd) * DH_ + e];
    }
    qo[rbase + e] = (_Float16)(aq * qscale);
    ko[rbase + e] = (_Float16)ak;
    vT[vbase + (size_t)e * SPAD] = (_Float16)av;
  }
}

// ---------------- Kernel 4: flash attention via v_wmma_f32_16x16x32_f16 ----------------
// One wave per (b, h, 16-row q tile). QK^T: K=DH(8) zero-padded to 32.
// Tile-wide scalar running max (exact flash algebra, wave-uniform alpha),
// row-sums computed by WMMA against a ones matrix (same C-layout as O/lrow),
// P moved C-layout -> A-layout through LDS with packed b32 stores.
template <bool MASK>
__device__ __forceinline__ void attn_chunk(int kc,
                                           const _Float16* __restrict__ kbp,
                                           const _Float16* __restrict__ vbp,
                                           const _Float16* __restrict__ zpage,
                                           const v16h& aq, const v16h& ones,
                                           int nn, int hi,
                                           _Float16* P, float& mrun, float* lrow, v8f& o) {
  v8f s01[2];
#pragma unroll
  for (int t = 0; t < 2; ++t) {
    // B operand (K^T): lanes 0-15 hold column N=nn, K rows 0..15 (dh, upper 8 = pad);
    // lanes 16-31 are the zero-padded K=16..31 region -> zero page.
    const _Float16* bkp = (hi == 0) ? (kbp + (size_t)(kc + t * 16 + nn) * QKSTRIDE) : zpage;
    v16h bk_ = *(const v16h*)bkp;
    v8f z = {};
    s01[t] = __builtin_amdgcn_wmma_f32_16x16x32_f16(false, aq, false, bk_, (short)0, z, false, false);
  }

  // tile-wide max: 15 in-register max + 5 xor-shuffles over the full wave
  float mt = -1e30f;
#pragma unroll
  for (int i = 0; i < 8; ++i) {
    float a0 = s01[0][i];
    float a1 = s01[1][i];
    if (MASK) {
      if (kc + nn >= S_)      a0 = -1e30f;
      if (kc + 16 + nn >= S_) a1 = -1e30f;
      s01[0][i] = a0; s01[1][i] = a1;
    }
    mt = fmaxf(mt, fmaxf(a0, a1));
  }
#pragma unroll
  for (int off = 1; off <= 16; off <<= 1) mt = fmaxf(mt, __shfl_xor(mt, off, 32));

  float mnew = fmaxf(mrun, mt);
  float alpha = __expf(mrun - mnew);
  mrun = mnew;

  // exp + packed f16 store into LDS (permuted columns: key kk -> kpos(kk))
#pragma unroll
  for (int i = 0; i < 8; ++i) {
    float p0 = __expf(s01[0][i] - mnew);
    float p1 = __expf(s01[1][i] - mnew);
    v2fp pk = __builtin_amdgcn_cvt_pkrtz(p0, p1);  // cols kpos(nn)=2nn, kpos(16+nn)=2nn+1
    *(v2fp*)&P[(hi * 8 + i) * 32 + 2 * nn] = pk;
  }
  __syncthreads();
  // A-layout read-back: lane<16 -> row nn, permuted cols 0..15; lane>=16 -> 16..31
  v16h ap = *(const v16h*)&P[nn * 32 + hi * 16];
  __syncthreads();   // protects next chunk's P stores (WAR)

  // row-sums via WMMA against ones: C layout matches lrow/o row indexing exactly
  v8f z = {};
  v8f rsum = __builtin_amdgcn_wmma_f32_16x16x32_f16(false, ap, false, ones, (short)0, z, false, false);

  // B operand = V (pre-permuted): column N = nn (dh, valid < 8), K = permuted key
  const _Float16* bvp = (nn < DH_) ? (vbp + (size_t)nn * SPAD + kc + hi * 16) : zpage;
  v16h bv_ = *(const v16h*)bvp;

#pragma unroll
  for (int i = 0; i < 8; ++i) {
    lrow[i] = lrow[i] * alpha + rsum[i];
    o[i] *= alpha;
  }
  o = __builtin_amdgcn_wmma_f32_16x16x32_f16(false, ap, false, bv_, (short)0, o, false, false);
}

__global__ void __launch_bounds__(32) k_attn(const _Float16* __restrict__ q,
                                             const _Float16* __restrict__ k,
                                             const _Float16* __restrict__ vT,
                                             const _Float16* __restrict__ zpage,
                                             const float* __restrict__ tokens,
                                             float* __restrict__ out) {
  int qt = blockIdx.x;          // 0..64
  int h  = blockIdx.y;
  int b  = blockIdx.z;
  int lane = threadIdx.x;
  int nn = lane & 15;
  int hi = lane >> 4;           // half-wave select
  __shared__ __align__(32) _Float16 P[16 * 32];

  const _Float16* qbp = q  + ((size_t)(b * H_ + h) * SPAD) * QKSTRIDE;
  const _Float16* kbp = k  + ((size_t)(b * H_ + h) * SPAD) * QKSTRIDE;
  const _Float16* vbp = vT + ((size_t)(b * H_ + h) * DH_) * SPAD;

  // A operand for QK^T: lanes 0-15 hold row M=nn (K pad already zero in memory);
  // lanes 16-31 are the zero-padded K region -> zero page.
  const _Float16* aqp = (hi == 0) ? (qbp + (size_t)(qt * 16 + nn) * QKSTRIDE) : zpage;
  v16h aq = *(const v16h*)aqp;

  v16h ones;
#pragma unroll
  for (int j = 0; j < 16; ++j) ones[j] = (_Float16)1.0f;

  v8f o = {};
  float mrun = -1e30f;
  float lrow[8];
#pragma unroll
  for (int i = 0; i < 8; ++i) lrow[i] = 0.f;

#pragma unroll 1
  for (int c = 0; c < NCHUNK - 1; ++c)
    attn_chunk<false>(c * 32, kbp, vbp, zpage, aq, ones, nn, hi, P, mrun, lrow, o);
  attn_chunk<true>((NCHUNK - 1) * 32, kbp, vbp, zpage, aq, ones, nn, hi, P, mrun, lrow, o);

  // out = tokens + attn (residual). lane column = dh, rows hi*8+i
  if (nn < DH_) {
#pragma unroll
    for (int i = 0; i < 8; ++i) {
      int row = qt * 16 + hi * 8 + i;
      if (row < S_) {
        size_t oi = ((size_t)b * S_ + row) * D_ + h * DH_ + nn;
        out[oi] = tokens[oi] + o[i] / lrow[i];
      }
    }
  }
}

// ---------------- Kernel 5: LN2 row0 -> encoder MLP -> residual -> classifier softmax ----------------
__global__ void k_head(const float* __restrict__ out, const float* __restrict__ mu, const float* __restrict__ rs,
                       const float* __restrict__ g2, const float* __restrict__ be2,
                       const float* __restrict__ w_enc, const float* __restrict__ b_enc,
                       const float* __restrict__ w_out, const float* __restrict__ b_out,
                       float* __restrict__ y_out) {
  int b = blockIdx.x;
  int e = threadIdx.x;            // 64 threads
  __shared__ float xn[D_];
  __shared__ float hvec[D_];
  __shared__ float logits[OUTD];
  float m = mu[b], r = rs[b];
  float x0 = out[(size_t)b * S_ * D_ + e];      // row 0 (CLS)
  xn[e] = (x0 - m) * r * g2[e] + be2[e];        // g2[0][:], be2[0][:]
  __syncthreads();
  float acc = b_enc[e];
#pragma unroll 8
  for (int dd = 0; dd < D_; ++dd) acc += xn[dd] * w_enc[dd * D_ + e];
  hvec[e] = x0 + fmaxf(acc, 0.f);
  __syncthreads();
  if (e < OUTD) {
    float lg = b_out[e];
#pragma unroll 8
    for (int dd = 0; dd < D_; ++dd) lg += hvec[dd] * w_out[dd * OUTD + e];
    logits[e] = lg;
  }
  __syncthreads();
  if (e == 0) {
    float mx = -1e30f;
    for (int o2 = 0; o2 < OUTD; ++o2) mx = fmaxf(mx, logits[o2]);
    float s = 0.f;
    float ex[OUTD];
    for (int o2 = 0; o2 < OUTD; ++o2) { ex[o2] = __expf(logits[o2] - mx); s += ex[o2]; }
    for (int o2 = 0; o2 < OUTD; ++o2) y_out[b * OUTD + o2] = ex[o2] / s;
  }
}

extern "C" void kernel_launch(void* const* d_in, const int* in_sizes, int n_in,
                              void* d_out, int out_size, void* d_ws, size_t ws_size,
                              hipStream_t stream) {
  (void)in_sizes; (void)n_in; (void)out_size; (void)ws_size;
  const float* images = (const float*)d_in[0];
  const float* w_map  = (const float*)d_in[1];
  const float* b_map  = (const float*)d_in[2];
  const float* cls    = (const float*)d_in[3];
  const float* g1     = (const float*)d_in[4];
  const float* be1    = (const float*)d_in[5];
  const float* wq     = (const float*)d_in[6];
  const float* bq     = (const float*)d_in[7];
  const float* wk     = (const float*)d_in[8];
  const float* bk     = (const float*)d_in[9];
  const float* wv     = (const float*)d_in[10];
  const float* bv     = (const float*)d_in[11];
  const float* g2     = (const float*)d_in[12];
  const float* be2    = (const float*)d_in[13];
  const float* w_enc  = (const float*)d_in[14];
  const float* b_enc  = (const float*)d_in[15];
  const float* w_out  = (const float*)d_in[16];
  const float* b_out  = (const float*)d_in[17];

  char* ws = (char*)d_ws;
  auto carve = [&](size_t bytes) {
    char* p = ws;
    ws += (bytes + 255) & ~(size_t)255;
    return (void*)p;
  };
  _Float16* zpage  = (_Float16*)carve(256);
  float*    tokens = (float*)carve((size_t)B_ * S_ * D_ * sizeof(float));
  float*    outb   = (float*)carve((size_t)B_ * S_ * D_ * sizeof(float));
  _Float16* qb     = (_Float16*)carve((size_t)B_ * H_ * SPAD * QKSTRIDE * sizeof(_Float16));
  _Float16* kb     = (_Float16*)carve((size_t)B_ * H_ * SPAD * QKSTRIDE * sizeof(_Float16));
  _Float16* vTb    = (_Float16*)carve((size_t)B_ * H_ * DH_ * SPAD * sizeof(_Float16));
  float*    mu1    = (float*)carve(B_ * sizeof(float));
  float*    rs1    = (float*)carve(B_ * sizeof(float));
  float*    mu2    = (float*)carve(B_ * sizeof(float));
  float*    rs2    = (float*)carve(B_ * sizeof(float));

  k_zero<<<1, 128, 0, stream>>>(zpage);
  int n1 = B_ * S_ * D_;
  k_tokens<<<(n1 + 255) / 256, 256, 0, stream>>>(images, w_map, b_map, cls, tokens);
  k_stats<<<B_, 256, 0, stream>>>(tokens, mu1, rs1);
  int n3 = B_ * H_ * SPAD;
  k_qkv<<<(n3 + 255) / 256, 256, 0, stream>>>(tokens, mu1, rs1, g1, be1,
                                              wq, bq, wk, bk, wv, bv, qb, kb, vTb);
  dim3 ag(QTILES, H_, B_);
  k_attn<<<ag, 32, 0, stream>>>(qb, kb, vTb, zpage, tokens, outb);
  k_stats<<<B_, 256, 0, stream>>>(outb, mu2, rs2);
  k_head<<<B_, 64, 0, stream>>>(outb, mu2, rs2, g2, be2, w_enc, b_enc, w_out, b_out,
                                (float*)d_out);
}
